// Head_5059471474749
// MI455X (gfx1250) — matrix operand order repaired and checked by reading, not compiled, not run on hardware
//
#include <hip/hip_runtime.h>
#include <hip/hip_bf16.h>

// ---------------------------------------------------------------------------
// Fused causal single-head attention for MI455X (gfx1250, wave32, WMMA).
//   B=8, T=2048, E=1024, H=128
// Pipeline:
//   (1) pack_w:  Wq/Wk/Wv (f32, ExH) -> bf16 B-fragment layout in workspace
//   (2) proj:    xq/xk = X@W (bf16, row-major TxH);  xvT = (V@Wv)^T (bf16, HxT)
//                W fragments double-buffered in LDS via async-to-LDS, shared
//                by all 4 waves (DMA for chunk c+1 overlaps WMMA on chunk c).
//   (3) attn:    flash-attention, 16 query rows / wave, 32 keys / iteration.
//                K/V tiles double-buffered in LDS via async-to-LDS, shared by
//                4 waves -> 4x less L2 traffic + DMA/compute overlap.
// LDS rows are padded (K:272B, V/P:80B, W: lane-skew) to kill bank conflicts.
// Workspace: 768KB packed weights + 3 x 4MB bf16 activations (~13MB).
// ---------------------------------------------------------------------------

#define Bsz 8
#define Tsz 2048
#define Esz 1024
#define Hsz 128
#define NT  (Bsz * Tsz)          // 16384 total rows

#define KROW 136                 // Kbuf row stride (ushorts): 128 data + 8 pad
#define VROW 40                  // Vbuf row stride (ushorts): 32 data + 8 pad
#define PROW 40                  // Plds row stride (ushorts): 32 data + 8 pad
#define WFRAG 272                // Wlds frag stride (dwords): 256 data + skew pad

typedef __attribute__((ext_vector_type(16))) __bf16          v16bf;
typedef __attribute__((ext_vector_type(2)))  __bf16          v2bf;
typedef __attribute__((ext_vector_type(16))) unsigned short  v16u;
typedef __attribute__((ext_vector_type(8)))  unsigned short  v8u;
typedef __attribute__((ext_vector_type(8)))  unsigned int    v8ui;
typedef __attribute__((ext_vector_type(4)))  int             v4i;
typedef __attribute__((ext_vector_type(8)))  float           v8f;

#define GLOBAL_AS __attribute__((address_space(1)))
#define LDS_AS    __attribute__((address_space(3)))

#if defined(__gfx1250__) && __has_builtin(__builtin_amdgcn_global_load_async_to_lds_b128)
#define HAVE_ASYNC_LDS 1
#else
#define HAVE_ASYNC_LDS 0
#endif

__device__ __forceinline__ unsigned short f2bf(float f) {
    unsigned int u = __builtin_bit_cast(unsigned int, f);
    unsigned int r = u + 0x7FFFu + ((u >> 16) & 1u);   // round-to-nearest-even
    return (unsigned short)(r >> 16);
}

// packed pair conversion: (a,b) -> u32 {lo=bf16(a), hi=bf16(b)}
__device__ __forceinline__ unsigned int f2bf_pk(float a, float b) {
#if defined(__gfx1250__) && __has_builtin(__builtin_amdgcn_cvt_pk_bf16_f32)
    v2bf r = __builtin_amdgcn_cvt_pk_bf16_f32(a, b);
    return __builtin_bit_cast(unsigned int, r);
#else
    return (unsigned int)f2bf(a) | ((unsigned int)f2bf(b) << 16);
#endif
}

__device__ __forceinline__ v8f wmma_bf16(v16u a, v16u b, v8f c) {
    return __builtin_amdgcn_wmma_f32_16x16x32_bf16(
        false, __builtin_bit_cast(v16bf, a),
        false, __builtin_bit_cast(v16bf, b),
        (short)0, c, false, false);
}

__device__ __forceinline__ v16u cat8(v8u lo, v8u hi) {
    return __builtin_shufflevector(lo, hi, 0,1,2,3,4,5,6,7,8,9,10,11,12,13,14,15);
}

// 16B global -> LDS copy; async (ASYNCcnt) path when the builtin exists.
// Builtin signature (from hipcc diagnostic): (v4i __device__*, v4i __shared__*, imm, imm)
__device__ __forceinline__ void stage_b128(const void* g, void* l) {
#if HAVE_ASYNC_LDS
    __builtin_amdgcn_global_load_async_to_lds_b128(
        (GLOBAL_AS v4i*)g, (LDS_AS v4i*)l, 0, 0);
#else
    *(uint4*)l = *(const uint4*)g;
#endif
}

// Wait until at most N async-to-LDS ops remain outstanding (in-order retire).
template <int N>
__device__ __forceinline__ void wait_async() {
#if HAVE_ASYNC_LDS
#if __has_builtin(__builtin_amdgcn_s_wait_asynccnt)
    __builtin_amdgcn_s_wait_asynccnt(N);
#else
    asm volatile("s_wait_asynccnt %0" :: "i"(N) : "memory");
#endif
#endif
}

// ---------------------------------------------------------------------------
// (1) Pack W (E x H, f32, row-major) into bf16 B-fragment layout used here:
//     wpack[w][chunk(32)][tile(8)][lane(32)][vgpr(8)]  (u32 = bf16 pair)
//     B-frag (32x16), contiguous k order, same as A: lane L -> col h = tile*16 + L%16,
//     K rows = chunk*32 + (L/16)*16 + 2v + {0,1}
// ---------------------------------------------------------------------------
__global__ void pack_w(const float* __restrict__ Wq, const float* __restrict__ Wk,
                       const float* __restrict__ Wv, unsigned int* __restrict__ wpack) {
    int idx = blockIdx.x * blockDim.x + threadIdx.x;     // 3*32*8*32*8 = 196608
    if (idx >= 3 * 32 * 8 * 32 * 8) return;
    int v     =  idx        & 7;
    int lane  = (idx >> 3)  & 31;
    int tile  = (idx >> 8)  & 7;
    int chunk = (idx >> 11) & 31;
    int w     =  idx >> 16;
    const float* W = (w == 0) ? Wq : (w == 1) ? Wk : Wv;
    int h = tile * 16 + (lane & 15);
    int k = chunk * 32 + ((lane >> 4) << 4) + 2 * v;
    wpack[idx] = f2bf_pk(W[k * Hsz + h], W[(k + 1) * Hsz + h]);
}

// ---------------------------------------------------------------------------
// (2) Projections. One wave = 16 output rows x 128 cols (8 tiles), K=1024 in
//     32 chunks of 32.  A-frag from f32 input (HBM); B-frags double-buffered
//     in LDS (async DMA for chunk c+1 overlaps the 8 WMMAs of chunk c).
//     blockIdx.y selects 0:q->xq  1:k->xk  2:v->xvT(transposed store).
// ---------------------------------------------------------------------------
__global__ void __launch_bounds__(128) proj_kernel(
        const float* __restrict__ k_in, const float* __restrict__ q_in,
        const float* __restrict__ v_in, const unsigned int* __restrict__ wpack,
        unsigned short* __restrict__ xq, unsigned short* __restrict__ xk,
        unsigned short* __restrict__ xvT) {
    __shared__ unsigned int Wlds[2][8 * WFRAG];          // 2 x 8.5KB, bank-skewed
    const int tid  = threadIdx.x;
    const int lane = tid & 31;
    const int wid  = tid >> 5;
    const int w    = blockIdx.y;
    const int row0 = (blockIdx.x * 4 + wid) * 16;        // never crosses a batch
    const float* src = (w == 0) ? q_in : (w == 1) ? k_in : v_in;

    // stage one chunk's 8 B-frags (8KB) into buffer `buf`, bank-skewed
    auto stage_chunk = [&](int cc, int buf) {
        const unsigned int* wsrc = wpack + ((w * 32 + cc) << 11);
#pragma unroll
        for (int i = 0; i < 4; ++i) {
            int g    = tid + i * 128;                    // 512 x 16B chunks
            int t    = g >> 6;
            int rem  = g & 63;
            int l    = rem >> 1;
            int half = rem & 1;
            stage_b128(wsrc + t * 256 + l * 8 + half * 4,
                       &Wlds[buf][t * WFRAG + l * 8 + ((l >> 3) << 2) + half * 4]);
        }
    };

    v8f acc[8] = {};
    const int arow = row0 + (lane & 15);
    const int akb  = (lane >> 4) << 4;                   // 0 or 16
    const int woff = lane * 8 + ((lane >> 3) << 2);      // skewed lane offset

    stage_chunk(0, 0);                                   // prologue
    for (int c = 0; c < 32; ++c) {
        const int cur = c & 1;
        if (c + 1 < 32) { stage_chunk(c + 1, cur ^ 1); wait_async<4>(); }
        else             { wait_async<0>(); }
        __syncthreads();

        // ---- A-frag: elements 0..15 = K kb..kb+15 ----
        const float* ap = src + (long)arow * Esz + c * 32 + akb;
        v8ui au;
#pragma unroll
        for (int j = 0; j < 4; ++j) {
            au[j]     = f2bf_pk(ap[2 * j],      ap[2 * j + 1]);
            au[j + 4] = f2bf_pk(ap[8 + 2 * j], ap[9 + 2 * j]);
        }
        v16u a = __builtin_bit_cast(v16u, au);

#pragma unroll
        for (int t = 0; t < 8; ++t) {
            const unsigned int* Wb = &Wlds[cur][t * WFRAG + woff];
            v16u bfrag = cat8(*(const v8u*)Wb, *(const v8u*)(Wb + 4));
            acc[t] = wmma_bf16(a, bfrag, acc[t]);
        }
        __syncthreads();                                 // reads done before overwrite
    }

    // C layout: VGPR r -> row r + (lane/16)*8, col = lane%16
    const int rbase = row0 + ((lane >> 4) << 3);
#pragma unroll
    for (int t = 0; t < 8; ++t) {
        int h = t * 16 + (lane & 15);
#pragma unroll
        for (int r = 0; r < 8; ++r) {
            int row = rbase + r;
            unsigned short bv = f2bf(acc[t][r]);
            if (w == 0)       xq[row * Hsz + h] = bv;
            else if (w == 1)  xk[row * Hsz + h] = bv;
            else {
                int b  = row / Tsz, tt = row % Tsz;
                xvT[((b * Hsz + h) * Tsz) + tt] = bv;    // transposed for PV B-frags
            }
        }
    }
}

// ---------------------------------------------------------------------------
// (3) Flash attention.  4 waves/block, each wave = 16 query rows, full H=128
//     accumulator.  32 keys/iteration; K (32x128) and V^T (128x32) tiles are
//     double-buffered in LDS (async DMA overlaps WMMA) and shared by 4 waves.
// ---------------------------------------------------------------------------
__global__ void __launch_bounds__(128) attn_kernel(
        const unsigned short* __restrict__ xq, const unsigned short* __restrict__ xk,
        const unsigned short* __restrict__ xvT, float* __restrict__ out) {
    __shared__ unsigned short Kbuf[2][32 * KROW];        // 2 x 8.5KB
    __shared__ unsigned short Vbuf[2][128 * VROW];       // 2 x 10KB
    __shared__ unsigned short Plds[4][16 * PROW];        // 1.25KB per wave

    const int tid  = threadIdx.x;
    const int lane = tid & 31;
    const int wid  = tid >> 5;
    const int tile0 = blockIdx.x * 4;                    // 4 consecutive query tiles
    const int b  = tile0 / (Tsz / 16);                   // same batch for all 4
    const int i0 = ((tile0 % (Tsz / 16)) + wid) * 16;    // this wave's query rows
    const int i0max = ((tile0 % (Tsz / 16)) + 3) * 16;   // block-uniform trip count

    const unsigned short* xqB = xq  + (long)b * Tsz * Hsz;
    const unsigned short* xkB = xk  + (long)b * Tsz * Hsz;
    const unsigned short* xvB = xvT + (long)b * Hsz * Tsz;

    const int n  = lane & 15;
    const int hg = lane >> 4;                            // lane-group (0/1)

    // stage K (32x128) and V^T (128x32) tiles for keys [j0s, j0s+32)
    auto stage_tiles = [&](int j0s, int buf) {
        const unsigned short* ksrc = xkB + j0s * Hsz;
#pragma unroll
        for (int i = 0; i < 4; ++i) {
            int g = tid + i * 128;                       // 512 x 16B chunks
            stage_b128(ksrc + g * 8, &Kbuf[buf][(g >> 4) * KROW + ((g & 15) << 3)]);
        }
        const unsigned short* vsrc = xvB + tid * Tsz + j0s;
#pragma unroll
        for (int i = 0; i < 4; ++i)
            stage_b128(vsrc + i * 8, &Vbuf[buf][tid * VROW + i * 8]);
    };

    // Q A-fragments: 4 chunks of K=32 over H=128
    v16u qa[4];
    {
        const unsigned short* qp = xqB + (i0 + n) * Hsz + (hg << 3);
#pragma unroll
        for (int c = 0; c < 4; ++c)
            qa[c] = cat8(*(const v8u*)(qp + c * 32), *(const v8u*)(qp + c * 32 + 16));
    }

    float mrow[8], lrow[8];
#pragma unroll
    for (int r = 0; r < 8; ++r) { mrow[r] = -3.0e38f; lrow[r] = 0.0f; }
    v8f acc[8] = {};

    const float scale = 0.03125f;                        // 1/sqrt(E) = 1/32
    const int rbase = i0 + (hg << 3);                    // this lane's first row
    const int nIter = i0max / 32 + 1;                    // j0 <= T-32 always

    stage_tiles(0, 0);                                   // prologue
    for (int it = 0; it < nIter; ++it) {
        const int j0  = it * 32;
        const int cur = it & 1;
        if (it + 1 < nIter) { stage_tiles(j0 + 32, cur ^ 1); wait_async<8>(); }
        else                 { wait_async<0>(); }
        __syncthreads();

        if (j0 <= i0) {                                  // wave-uniform guard
            // ---- S = scale * Q K^T, two 16-key halves ----
            v8f s[2];
#pragma unroll
            for (int half = 0; half < 2; ++half) {
                s[half] = (v8f)0.0f;
                const unsigned short* kp = &Kbuf[cur][(half * 16 + n) * KROW + (hg << 3)];
#pragma unroll
                for (int c = 0; c < 4; ++c) {
                    v16u kf = cat8(*(const v8u*)(kp + c * 32),
                                   *(const v8u*)(kp + c * 32 + 16));
                    s[half] = wmma_bf16(qa[c], kf, s[half]);
                }
            }
            // ---- scale + causal mask ----
#pragma unroll
            for (int r = 0; r < 8; ++r) {
                int row = rbase + r;
#pragma unroll
                for (int half = 0; half < 2; ++half) {
                    int col = j0 + half * 16 + n;
                    float x = s[half][r] * scale;
                    s[half][r] = (col > row) ? -3.0e38f : x;
                }
            }
            // ---- online softmax (16-lane butterflies stay in half-wave) ----
            float alph[8];
#pragma unroll
            for (int r = 0; r < 8; ++r) {
                float vmax = fmaxf(s[0][r], s[1][r]);
                vmax = fmaxf(vmax, __shfl_xor(vmax, 1, 32));
                vmax = fmaxf(vmax, __shfl_xor(vmax, 2, 32));
                vmax = fmaxf(vmax, __shfl_xor(vmax, 4, 32));
                vmax = fmaxf(vmax, __shfl_xor(vmax, 8, 32));
                float mnew  = fmaxf(mrow[r], vmax);
                float alpha = __expf(mrow[r] - mnew);
                float p0 = __expf(s[0][r] - mnew);
                float p1 = __expf(s[1][r] - mnew);
                float rs = p0 + p1;
                rs += __shfl_xor(rs, 1, 32);
                rs += __shfl_xor(rs, 2, 32);
                rs += __shfl_xor(rs, 4, 32);
                rs += __shfl_xor(rs, 8, 32);
                lrow[r] = lrow[r] * alpha + rs;
                mrow[r] = mnew;
                alph[r] = alpha;
                s[0][r] = p0; s[1][r] = p1;
            }
#pragma unroll
            for (int t = 0; t < 8; ++t)
#pragma unroll
                for (int r = 0; r < 8; ++r) acc[t][r] *= alph[r];

            // ---- P (C layout) -> LDS -> A-frag (16x32 bf16) ----
            unsigned short* Pw = &Plds[wid][0];
#pragma unroll
            for (int r = 0; r < 8; ++r) {
                int prow = r + (hg << 3);
                Pw[prow * PROW + n]      = f2bf(s[0][r]);
                Pw[prow * PROW + 16 + n] = f2bf(s[1][r]);
            }
            v16u pa;
            {
                const unsigned short* pp = Pw + n * PROW + (hg << 3);
                pa = cat8(*(const v8u*)pp, *(const v8u*)(pp + 16));
            }
            // ---- acc += P @ V ----
#pragma unroll
            for (int t = 0; t < 8; ++t) {
                const unsigned short* vp = &Vbuf[cur][(t * 16 + n) * VROW + (hg << 3)];
                v16u vfrag = cat8(*(const v8u*)vp, *(const v8u*)(vp + 16));
                acc[t] = wmma_bf16(pa, vfrag, acc[t]);
            }
        }
        __syncthreads();                                 // reads done before overwrite
    }

    // ---- epilogue: out[b, row, h] = acc / l ----
    float inv[8];
#pragma unroll
    for (int r = 0; r < 8; ++r) inv[r] = 1.0f / lrow[r];
#pragma unroll
    for (int t = 0; t < 8; ++t) {
        int h = t * 16 + n;
#pragma unroll
        for (int r = 0; r < 8; ++r) {
            int row = rbase + r;                         // row within batch
            out[((long)b * Tsz + row) * Hsz + h] = acc[t][r] * inv[r];
        }
    }
}

// ---------------------------------------------------------------------------
extern "C" void kernel_launch(void* const* d_in, const int* in_sizes, int n_in,
                              void* d_out, int out_size, void* d_ws, size_t ws_size,
                              hipStream_t stream) {
    const float* k_in = (const float*)d_in[0];
    const float* q_in = (const float*)d_in[1];
    const float* v_in = (const float*)d_in[2];
    // d_in[3] = causal mask (recomputed analytically; unused)
    const float* Wk = (const float*)d_in[4];
    const float* Wq = (const float*)d_in[5];
    const float* Wv = (const float*)d_in[6];
    float* out = (float*)d_out;

    char* ws = (char*)d_ws;
    unsigned int*   wpack = (unsigned int*)ws;                    // 768 KB
    unsigned short* xq    = (unsigned short*)(ws + (1u << 20));   // 4 MB
    unsigned short* xk    = xq + (size_t)NT * Hsz;                // 4 MB
    unsigned short* xvT   = xk + (size_t)NT * Hsz;                // 4 MB

    pack_w<<<768, 256, 0, stream>>>(Wq, Wk, Wv, wpack);

    dim3 pgrid(NT / (16 * 4), 3);                                 // (256, 3)
    proj_kernel<<<pgrid, 128, 0, stream>>>(k_in, q_in, v_in, wpack, xq, xk, xvT);

    attn_kernel<<<NT / (16 * 4), 128, 0, stream>>>(xq, xk, xvT, out);
}
